// GraphMLPMixer4TreeNeighbour_30107720744963
// MI455X (gfx1250) — matrix operand order repaired and checked by reading, not compiled, not run on hardware
//
#include <hip/hip_runtime.h>
#include <hip/hip_bf16.h>

// ---------------- problem constants (from reference) ----------------
#define N_NODESC  150000
#define N_SUBC    300000
#define N_EDGESC  1200000
#define BATCHC    256
#define PATCHC    32
#define SEGC      (BATCHC*PATCHC)   // 8192
#define DIMC      128
#define NOUTC     64
#define EPSBN     1e-5f

typedef __attribute__((ext_vector_type(16))) _Float16 v16h;
typedef __attribute__((ext_vector_type(8)))  float    v8f;

union FragH { v16h v; uint4 q[2]; };
union FragC { v8f  v; float f[8]; };

__device__ __forceinline__ float geluf(float x) {
  float x3 = x*x*x;
  return 0.5f*x*(1.0f + tanhf(0.7978845608028654f*(x + 0.044715f*x3)));
}

// ---------------- generic WMMA f16 GEMM:  C = act(A @ Bpacked + bias) [+resid] ----------------
// A: [Mr,K] f16 row-major, Bp: packed per-lane fragments, Mr multiple of 16.
// Epilogue fully specialized at compile time: no branches around stores.
// ACT: 0=none, 1=relu, 2=gelu.
template<int K, int ACT, bool HAS_RESID, bool OUT_F32, bool OUT_F16>
__global__ void k_gemm_f16(const _Float16* __restrict__ A,
                           const _Float16* __restrict__ Bp,
                           const float* __restrict__ bias,
                           const float* __restrict__ resid,
                           float* __restrict__ Cf,
                           _Float16* __restrict__ Ch,
                           int N)
{
  constexpr int NK = K/32;
  const int lane  = threadIdx.x & 31;
  const int wave  = threadIdx.x >> 5;
  const int rt    = blockIdx.x;
  const int khalf = lane >> 4;
  const int row   = rt*16 + (lane & 15);

  // A fragments for this 16-row tile (ISA 16-bit A 16x32 layout):
  // elements 0..7  -> K = kb*32 + khalf*8 + 0..7
  // elements 8..15 -> K = kb*32 + 16 + khalf*8 + 0..7
  FragH afrag[NK];
#pragma unroll
  for (int kb = 0; kb < NK; ++kb) {
    const _Float16* pa = A + (size_t)row*K + kb*32 + khalf*8;
    afrag[kb].q[0] = *(const uint4*)(pa);
    afrag[kb].q[1] = *(const uint4*)(pa + 16);
  }

  const int ntiles = N >> 4;
  for (int ct = wave; ct < ntiles; ct += 4) {
    const int col = ct*16 + (lane & 15);
    const float bv = bias[col];          // all 8 accum elements of this lane share `col`
    FragC c;
#pragma unroll
    for (int r = 0; r < 8; ++r) c.f[r] = bv;
#pragma unroll
    for (int kb = 0; kb < NK; ++kb) {
      FragH b;
      const _Float16* pb = Bp + ((size_t)(ct*NK + kb)*32 + lane)*16;
      b.q[0] = *(const uint4*)(pb);
      b.q[1] = *(const uint4*)(pb + 8);
      c.v = __builtin_amdgcn_wmma_f32_16x16x32_f16(false, afrag[kb].v,
                                                   false, b.v,
                                                   (short)0, c.v, false, false);
    }
#pragma unroll
    for (int r = 0; r < 8; ++r) {
      const int orow = rt*16 + r + 8*khalf;           // C/D layout: VGPR r, lanes>=16 -> M+8
      float v = c.f[r];
      if constexpr (ACT == 1) v = fmaxf(v, 0.0f);
      if constexpr (ACT == 2) v = geluf(v);
      const size_t idx = (size_t)orow*N + col;
      if constexpr (HAS_RESID) v += resid[idx];
      if constexpr (OUT_F32) Cf[idx] = v;
      if constexpr (OUT_F16) Ch[idx] = (_Float16)v;
    }
  }
}

// pack f32 weight [K,N] into per-lane WMMA B fragments (16 f16 contiguous per lane)
__global__ void k_pack_w(const float* __restrict__ W, _Float16* __restrict__ out, int K, int N)
{
  int t = blockIdx.x*blockDim.x + threadIdx.x;
  int nk = K >> 5;
  int total = (N >> 4) * nk * 32;
  if (t >= total) return;
  int lane = t & 31;
  int kb = (t >> 5) % nk;
  int nt = t / (32*nk);
  int n = nt*16 + (lane & 15);
  int kbase = kb*32 + (lane >> 4)*8;
  _Float16* dst = out + (size_t)t*16;
  for (int i = 0; i < 16; ++i) dst[i] = (_Float16)W[(size_t)(kbase+i+(i&8))*N + n];
}

// ---------------- elementwise / gather / scatter helpers ----------------
__global__ void k_zero(float* p, size_t n) {
  for (size_t i = (size_t)blockIdx.x*blockDim.x + threadIdx.x; i < n; i += (size_t)gridDim.x*blockDim.x)
    p[i] = 0.0f;
}
__global__ void k_fill_i32(int* p, int v, int n) {
  int g = blockIdx.x*blockDim.x + threadIdx.x;
  if (g < n) p[g] = v;
}
__global__ void k_embed(const int* __restrict__ xt, const int* __restrict__ nmap,
                        const float* __restrict__ kemb, const float* __restrict__ vemb,
                        float* __restrict__ x)
{
  int g = blockIdx.x*blockDim.x + threadIdx.x;
  if (g >= N_SUBC*32) return;
  int m = g >> 5, lane = g & 31;
  int n = nmap[m];
  int t0 = xt[2*n], t1 = xt[2*n+1];
  float4 a = ((const float4*)(kemb + (size_t)t0*DIMC))[lane];
  float4 b = ((const float4*)(vemb + (size_t)t1*DIMC))[lane];
  float4 o; o.x=a.x+b.x; o.y=a.y+b.y; o.z=a.z+b.z; o.w=a.w+b.w;
  ((float4*)(x + (size_t)m*DIMC))[lane] = o;
}
__global__ void k_edge(const int* __restrict__ src, const int* __restrict__ dst,
                       const int* __restrict__ eattr, const int* __restrict__ emap,
                       const float* __restrict__ eemb, const float* __restrict__ x,
                       float* __restrict__ agg)
{
  int g = blockIdx.x*blockDim.x + threadIdx.x;
  if (g >= N_EDGESC*32) return;
  int e = g >> 5, lane = g & 31;
  int s = src[e], d = dst[e];
  int ea = eattr[emap[e]];
  float4 xv = ((const float4*)(x + (size_t)s*DIMC))[lane];
  float4 ev = ((const float4*)(eemb + (size_t)ea*DIMC))[lane];
  float* ag = agg + (size_t)d*DIMC + lane*4;
  atomicAdd(ag+0, fmaxf(xv.x+ev.x, 0.f));
  atomicAdd(ag+1, fmaxf(xv.y+ev.y, 0.f));
  atomicAdd(ag+2, fmaxf(xv.z+ev.z, 0.f));
  atomicAdd(ag+3, fmaxf(xv.w+ev.w, 0.f));
}
__global__ void k_seg_sum(const float* __restrict__ x, const int* __restrict__ idx,
                          float* __restrict__ sum, float* __restrict__ cnt, int rows)
{
  int g = blockIdx.x*blockDim.x + threadIdx.x;
  if (g >= rows*32) return;
  int m = g >> 5, lane = g & 31;
  int s = idx[m];
  float4 v = ((const float4*)(x + (size_t)m*DIMC))[lane];
  float* dp = sum + (size_t)s*DIMC + lane*4;
  atomicAdd(dp+0, v.x); atomicAdd(dp+1, v.y); atomicAdd(dp+2, v.z); atomicAdd(dp+3, v.w);
  if (lane == 0) atomicAdd(&cnt[s], 1.0f);
}
__global__ void k_divcnt(float* __restrict__ buf, const float* __restrict__ cnt, int rows)
{
  int g = blockIdx.x*blockDim.x + threadIdx.x;
  if (g >= rows*DIMC) return;
  buf[g] /= fmaxf(cnt[g >> 7], 1.0f);
}
__global__ void k_gather(const float* __restrict__ in, const int* __restrict__ idx,
                         float* __restrict__ out, int rows)
{
  int g = blockIdx.x*blockDim.x + threadIdx.x;
  if (g >= rows*32) return;
  int m = g >> 5, lane = g & 31;
  ((float4*)(out + (size_t)m*DIMC))[lane] = ((const float4*)(in + (size_t)idx[m]*DIMC))[lane];
}
__global__ void k_gather_h(const float* __restrict__ in, const int* __restrict__ idx,
                           _Float16* __restrict__ out, int rows)
{
  int g = blockIdx.x*blockDim.x + threadIdx.x;
  if (g >= rows*32) return;
  int m = g >> 5, lane = g & 31;
  float4 v = ((const float4*)(in + (size_t)idx[m]*DIMC))[lane];
  _Float16* o = out + (size_t)m*DIMC + lane*4;
  o[0]=(_Float16)v.x; o[1]=(_Float16)v.y; o[2]=(_Float16)v.z; o[3]=(_Float16)v.w;
}
__global__ void k_add_h(const float* __restrict__ a, const float* __restrict__ b,
                        _Float16* __restrict__ o, size_t n)
{
  size_t g = (size_t)blockIdx.x*blockDim.x + threadIdx.x;
  if (g >= n) return;
  o[g] = (_Float16)(a[g] + b[g]);
}
// ---------------- BN ----------------
__global__ void k_bn_stats(const float* __restrict__ y, float* __restrict__ st,
                           int Mr, int C, int rpb)
{
  int c = threadIdx.x;                // blockDim.x == C
  int r0 = blockIdx.x*rpb, r1 = min(r0 + rpb, Mr);
  float s = 0.f, sq = 0.f;
  for (int r = r0; r < r1; ++r) { float v = y[(size_t)r*C + c]; s += v; sq += v*v; }
  atomicAdd(&st[c], s); atomicAdd(&st[C+c], sq);
}
__global__ void k_bn_relu_add(const float* __restrict__ y, const float* __restrict__ st,
                              const float* __restrict__ g_, const float* __restrict__ b_,
                              float* __restrict__ x, int Mr, int C)
{
  size_t g = (size_t)blockIdx.x*blockDim.x + threadIdx.x;
  if (g >= (size_t)Mr*C) return;
  int c = (int)(g & (size_t)(C-1));
  float mean = st[c]/Mr;
  float var  = st[C+c]/Mr - mean*mean;
  float v = g_[c]*(y[g]-mean)*rsqrtf(var+EPSBN) + b_[c];
  x[g] += fmaxf(v, 0.f);
}
__global__ void k_bn_relu_h(const float* __restrict__ y, const float* __restrict__ st,
                            const float* __restrict__ g_, const float* __restrict__ b_,
                            _Float16* __restrict__ o, int Mr, int C)
{
  size_t g = (size_t)blockIdx.x*blockDim.x + threadIdx.x;
  if (g >= (size_t)Mr*C) return;
  int c = (int)(g & (size_t)(C-1));
  float mean = st[c]/Mr;
  float var  = st[C+c]/Mr - mean*mean;
  float v = g_[c]*(y[g]-mean)*rsqrtf(var+EPSBN) + b_[c];
  o[g] = (_Float16)fmaxf(v, 0.f);
}
// ---------------- LN ----------------
__global__ void k_ln(const float* __restrict__ z, const float* __restrict__ g_,
                     const float* __restrict__ b_, float* __restrict__ of,
                     _Float16* __restrict__ oh)
{
  __shared__ float red[DIMC];
  int row = blockIdx.x, tid = threadIdx.x;
  float v = z[(size_t)row*DIMC + tid];
  red[tid] = v; __syncthreads();
  for (int s = DIMC/2; s > 0; s >>= 1) { if (tid < s) red[tid] += red[tid+s]; __syncthreads(); }
  float mean = red[0] / DIMC;
  __syncthreads();
  float dv = v - mean;
  red[tid] = dv*dv; __syncthreads();
  for (int s = DIMC/2; s > 0; s >>= 1) { if (tid < s) red[tid] += red[tid+s]; __syncthreads(); }
  float var = red[0] / DIMC;
  float o = g_[tid]*dv*rsqrtf(var+EPSBN) + b_[tid];
  if (of) of[(size_t)row*DIMC + tid] = o;
  if (oh) oh[(size_t)row*DIMC + tid] = (_Float16)o;
}
__global__ void k_zmask(const float* __restrict__ bm, const unsigned char* __restrict__ mask,
                        float* __restrict__ z)
{
  int g = blockIdx.x*blockDim.x + threadIdx.x;
  if (g >= SEGC*DIMC) return;
  int s = g >> 7;
  z[g] = mask[s] ? bm[g] : 0.0f;
}
// ---------------- token mixing (small matvecs; VALU) ----------------
__global__ void k_token_mix(const float* __restrict__ lnz, float* __restrict__ z,
                            const float* __restrict__ tW1, const float* __restrict__ tb1,
                            const float* __restrict__ tW2, const float* __restrict__ tb2)
{
  __shared__ float tile[PATCHC*DIMC];   // 16 KB
  __shared__ float Hh[64*DIMC];         // 32 KB
  int bb = blockIdx.x, tid = threadIdx.x;
  const float* src = lnz + (size_t)bb*PATCHC*DIMC;
  for (int i = tid; i < PATCHC*DIMC; i += 256) tile[i] = src[i];
  __syncthreads();
  for (int i = tid; i < 64*DIMC; i += 256) {
    int j = i >> 7, d = i & 127;
    float acc = tb1[j];
    for (int p = 0; p < PATCHC; ++p) acc += tile[p*DIMC + d] * tW1[p*64 + j];
    Hh[j*DIMC + d] = geluf(acc);
  }
  __syncthreads();
  for (int i = tid; i < PATCHC*DIMC; i += 256) {
    int p = i >> 7, d = i & 127;
    float acc = tb2[p];
    for (int j = 0; j < 64; ++j) acc += Hh[j*DIMC + d] * tW2[j*PATCHC + p];
    z[(size_t)bb*PATCHC*DIMC + i] += acc;
  }
}
// ---------------- root pooling (only the 256 needed segments of the N-node mean) ----------------
__global__ void k_rootset(const int* __restrict__ root_idx, int* __restrict__ map)
{
  int b = blockIdx.x*blockDim.x + threadIdx.x;
  if (b < BATCHC) map[root_idx[b]] = b;
}
__global__ void k_rootscatter(const float* __restrict__ x, const float* __restrict__ z,
                              const int* __restrict__ nmap, const int* __restrict__ bx,
                              const int* __restrict__ rootmap,
                              float* __restrict__ rsum, float* __restrict__ rcnt)
{
  int g = blockIdx.x*blockDim.x + threadIdx.x;
  if (g >= N_SUBC*32) return;
  int m = g >> 5, lane = g & 31;
  int b = rootmap[nmap[m]];
  if (b < 0) return;
  float4 xv = ((const float4*)(x + (size_t)m*DIMC))[lane];
  float* d0 = rsum + (size_t)b*(2*DIMC) + lane*4;
  atomicAdd(d0+0, xv.x); atomicAdd(d0+1, xv.y); atomicAdd(d0+2, xv.z); atomicAdd(d0+3, xv.w);
  float4 zv = ((const float4*)(z + (size_t)bx[m]*DIMC))[lane];
  float* d1 = rsum + (size_t)b*(2*DIMC) + DIMC + lane*4;
  atomicAdd(d1+0, zv.x); atomicAdd(d1+1, zv.y); atomicAdd(d1+2, zv.z); atomicAdd(d1+3, zv.w);
  if (lane == 0) atomicAdd(&rcnt[b], 1.0f);
}
__global__ void k_rootdiv_h(const float* __restrict__ rsum, const float* __restrict__ rcnt,
                            _Float16* __restrict__ o)
{
  int g = blockIdx.x*blockDim.x + threadIdx.x;
  if (g >= BATCHC*2*DIMC) return;
  int b = g / (2*DIMC);
  o[g] = (_Float16)(rsum[g] / fmaxf(rcnt[b], 1.0f));
}

// ====================================================================
extern "C" void kernel_launch(void* const* d_in, const int* in_sizes, int n_in,
                              void* d_out, int out_size, void* d_ws, size_t ws_size,
                              hipStream_t stream)
{
  // ---- inputs (setup_inputs dict order, params flattened depth-first in insertion order) ----
  const int* x_tokens     = (const int*)d_in[0];
  const int* edge_attr    = (const int*)d_in[1];
  const int* edge_index   = (const int*)d_in[2];
  const int* nodes_mapper = (const int*)d_in[3];
  const int* edges_mapper = (const int*)d_in[4];
  const int* batch_x      = (const int*)d_in[5];
  const int* root_idx     = (const int*)d_in[6];
  const unsigned char* mask = (const unsigned char*)d_in[7];
  const float* key_emb  = (const float*)d_in[8];
  const float* val_emb  = (const float*)d_in[9];
  const float* edge_emb = (const float*)d_in[10];
  const float* gW1[2] = {(const float*)d_in[11], (const float*)d_in[17]};
  const float* gb1[2] = {(const float*)d_in[12], (const float*)d_in[18]};
  const float* gW2[2] = {(const float*)d_in[13], (const float*)d_in[19]};
  const float* gb2[2] = {(const float*)d_in[14], (const float*)d_in[20]};
  const float* gbg[2] = {(const float*)d_in[15], (const float*)d_in[21]};
  const float* gbb[2] = {(const float*)d_in[16], (const float*)d_in[22]};
  const float* UW  = (const float*)d_in[23];
  const float* Ub  = (const float*)d_in[24];
  const float* Ubg = (const float*)d_in[25];
  const float* Ubb = (const float*)d_in[26];
  const int MIX0 = 27;
  const float* dW1 = (const float*)d_in[75];
  const float* db1 = (const float*)d_in[76];
  const float* dbg = (const float*)d_in[77];
  const float* dbb = (const float*)d_in[78];
  const float* dW2 = (const float*)d_in[79];
  const float* db2 = (const float*)d_in[80];

  // ---- workspace carve ----
  char* ws = (char*)d_ws;
  size_t off = 0;
  auto carve = [&](size_t bytes) -> char* {
    char* p = ws + off;
    off = (off + bytes + 255) & ~(size_t)255;
    return p;
  };
  float*    X    = (float*)carve((size_t)N_SUBC*DIMC*4);
  float*    TMP  = (float*)carve((size_t)N_SUBC*DIMC*4);
  _Float16* HA   = (_Float16*)carve((size_t)N_SUBC*DIMC*2);
  _Float16* HB   = (_Float16*)carve((size_t)N_SUBC*DIMC*2);
  float*    NSUM = (float*)carve((size_t)N_NODESC*DIMC*4);
  float*    NCNT = (float*)carve((size_t)N_NODESC*4);
  float*    BSUM = (float*)carve((size_t)SEGC*DIMC*4);
  float*    BCNT = (float*)carve((size_t)SEGC*4);
  float*    Z    = (float*)carve((size_t)SEGC*DIMC*4);
  float*    LNF  = (float*)carve((size_t)SEGC*DIMC*4);
  _Float16* LNH  = (_Float16*)carve((size_t)SEGC*DIMC*2);
  _Float16* G1H  = (_Float16*)carve((size_t)SEGC*4*DIMC*2);
  float*    STATS= (float*)carve(2*DIMC*4);
  float*    RSUM = (float*)carve((size_t)BATCHC*2*DIMC*4);
  float*    RCNT = (float*)carve((size_t)BATCHC*4);
  int*      RMAP = (int*)carve((size_t)N_NODESC*4);
  _Float16* RH   = (_Float16*)carve((size_t)BATCHC*2*DIMC*2);
  float*    DEC1 = (float*)carve((size_t)BATCHC*NOUTC*4);
  _Float16* D1H  = (_Float16*)carve((size_t)BATCHC*NOUTC*2);
  _Float16* W1p[2], *W2p[2], *Up, *C1p[4], *C2p[4], *DW1p, *DW2p;
  for (int l = 0; l < 2; ++l) { W1p[l] = (_Float16*)carve(128*128*2); W2p[l] = (_Float16*)carve(128*128*2); }
  Up = (_Float16*)carve(128*128*2);
  for (int l = 0; l < 4; ++l) { C1p[l] = (_Float16*)carve(128*512*2); C2p[l] = (_Float16*)carve(512*128*2); }
  DW1p = (_Float16*)carve(256*64*2);
  DW2p = (_Float16*)carve(64*64*2);

  auto pack = [&](const float* W, _Float16* dst, int K, int N) {
    int total = (N >> 4) * (K >> 5) * 32;
    k_pack_w<<<(total + 255)/256, 256, 0, stream>>>(W, dst, K, N);
  };
  auto zero = [&](float* p, size_t n) {
    int blocks = (int)(((n + 255)/256 < 8192) ? (n + 255)/256 : 8192);
    k_zero<<<blocks, 256, 0, stream>>>(p, n);
  };

  const int* esrc = edge_index;
  const int* edst = edge_index + N_EDGESC;
  const int GM = (N_SUBC*32 + 255)/256;        // per-(row,lane4) grids over M
  const int GMD = (int)(((size_t)N_SUBC*DIMC + 255)/256);

  // ---- pack all GEMM weights into WMMA fragment layout ----
  for (int l = 0; l < 2; ++l) { pack(gW1[l], W1p[l], 128, 128); pack(gW2[l], W2p[l], 128, 128); }
  pack(UW, Up, 128, 128);
  for (int l = 0; l < 4; ++l) {
    pack((const float*)d_in[MIX0 + 12*l + 8],  C1p[l], 128, 512);
    pack((const float*)d_in[MIX0 + 12*l + 10], C2p[l], 512, 128);
  }
  pack(dW1, DW1p, 256, 64);
  pack(dW2, DW2p, 64, 64);

  // ---- embed + gather subnodes ----
  k_embed<<<GM, 256, 0, stream>>>(x_tokens, nodes_mapper, key_emb, val_emb, X);

  // ---- GNN layers ----
  for (int l = 0; l < 2; ++l) {
    if (l > 0) {
      // subgraph pool -> U MLP -> BN+relu residual -> node-copy share
      zero(BSUM, (size_t)SEGC*DIMC); zero(BCNT, SEGC);
      k_seg_sum<<<GM, 256, 0, stream>>>(X, batch_x, BSUM, BCNT, N_SUBC);
      k_divcnt<<<(SEGC*DIMC + 255)/256, 256, 0, stream>>>(BSUM, BCNT, SEGC);
      k_gather_h<<<GM, 256, 0, stream>>>(BSUM, batch_x, HA, N_SUBC);
      k_gemm_f16<128,0,false,true,false><<<N_SUBC/16, 128, 0, stream>>>(HA, Up, Ub, nullptr, TMP, nullptr, 128);
      zero(STATS, 2*DIMC);
      k_bn_stats<<<(N_SUBC + 511)/512, DIMC, 0, stream>>>(TMP, STATS, N_SUBC, DIMC, 512);
      k_bn_relu_add<<<GMD, 256, 0, stream>>>(TMP, STATS, Ubg, Ubb, X, N_SUBC, DIMC);
      zero(NSUM, (size_t)N_NODESC*DIMC); zero(NCNT, N_NODESC);
      k_seg_sum<<<GM, 256, 0, stream>>>(X, nodes_mapper, NSUM, NCNT, N_SUBC);
      k_divcnt<<<(N_NODESC*DIMC + 255)/256, 256, 0, stream>>>(NSUM, NCNT, N_NODESC);
      k_gather<<<GM, 256, 0, stream>>>(NSUM, nodes_mapper, X, N_SUBC);
    }
    // GINE conv
    zero(TMP, (size_t)N_SUBC*DIMC);
    k_edge<<<(N_EDGESC*32 + 255)/256, 256, 0, stream>>>(esrc, edst, edge_attr, edges_mapper,
                                                        edge_emb, X, TMP);
    k_add_h<<<GMD, 256, 0, stream>>>(X, TMP, HA, (size_t)N_SUBC*DIMC);
    k_gemm_f16<128,1,false,false,true><<<N_SUBC/16, 128, 0, stream>>>(HA, W1p[l], gb1[l], nullptr, nullptr, HB, 128);
    k_gemm_f16<128,0,false,true,false><<<N_SUBC/16, 128, 0, stream>>>(HB, W2p[l], gb2[l], nullptr, TMP, nullptr, 128);
    zero(STATS, 2*DIMC);
    k_bn_stats<<<(N_SUBC + 511)/512, DIMC, 0, stream>>>(TMP, STATS, N_SUBC, DIMC, 512);
    k_bn_relu_add<<<GMD, 256, 0, stream>>>(TMP, STATS, gbg[l], gbb[l], X, N_SUBC, DIMC);
  }

  // ---- patch tokens z = masked seg_mean(x, batch_x) ----
  zero(BSUM, (size_t)SEGC*DIMC); zero(BCNT, SEGC);
  k_seg_sum<<<GM, 256, 0, stream>>>(X, batch_x, BSUM, BCNT, N_SUBC);
  k_divcnt<<<(SEGC*DIMC + 255)/256, 256, 0, stream>>>(BSUM, BCNT, SEGC);
  k_zmask<<<(SEGC*DIMC + 255)/256, 256, 0, stream>>>(BSUM, mask, Z);

  // ---- MLP-Mixer ----
  for (int l = 0; l < 4; ++l) {
    const float* ln1g = (const float*)d_in[MIX0 + 12*l + 0];
    const float* ln1b = (const float*)d_in[MIX0 + 12*l + 1];
    const float* tW1  = (const float*)d_in[MIX0 + 12*l + 2];
    const float* tb1  = (const float*)d_in[MIX0 + 12*l + 3];
    const float* tW2  = (const float*)d_in[MIX0 + 12*l + 4];
    const float* tb2  = (const float*)d_in[MIX0 + 12*l + 5];
    const float* ln2g = (const float*)d_in[MIX0 + 12*l + 6];
    const float* ln2b = (const float*)d_in[MIX0 + 12*l + 7];
    const float* cb1  = (const float*)d_in[MIX0 + 12*l + 9];
    const float* cb2  = (const float*)d_in[MIX0 + 12*l + 11];
    k_ln<<<SEGC, DIMC, 0, stream>>>(Z, ln1g, ln1b, LNF, nullptr);
    k_token_mix<<<BATCHC, 256, 0, stream>>>(LNF, Z, tW1, tb1, tW2, tb2);
    k_ln<<<SEGC, DIMC, 0, stream>>>(Z, ln2g, ln2b, nullptr, LNH);
    k_gemm_f16<128,2,false,false,true><<<SEGC/16, 128, 0, stream>>>(LNH, C1p[l], cb1, nullptr, nullptr, G1H, 512);
    k_gemm_f16<512,0,true,true,false><<<SEGC/16, 128, 0, stream>>>(G1H, C2p[l], cb2, Z, Z, nullptr, 128);
  }

  // ---- decoder: per-root mean of concat(x, mz[batch_x]) -> MLP ----
  k_fill_i32<<<(N_NODESC + 255)/256, 256, 0, stream>>>(RMAP, -1, N_NODESC);
  k_rootset<<<1, 256, 0, stream>>>(root_idx, RMAP);
  zero(RSUM, (size_t)BATCHC*2*DIMC); zero(RCNT, BATCHC);
  k_rootscatter<<<GM, 256, 0, stream>>>(X, Z, nodes_mapper, batch_x, RMAP, RSUM, RCNT);
  k_rootdiv_h<<<(BATCHC*2*DIMC + 255)/256, 256, 0, stream>>>(RSUM, RCNT, RH);
  k_gemm_f16<256,0,false,true,false><<<BATCHC/16, 128, 0, stream>>>(RH, DW1p, db1, nullptr, DEC1, nullptr, NOUTC);
  zero(STATS, 2*NOUTC);
  k_bn_stats<<<1, NOUTC, 0, stream>>>(DEC1, STATS, BATCHC, NOUTC, BATCHC);
  k_bn_relu_h<<<(BATCHC*NOUTC + 255)/256, 256, 0, stream>>>(DEC1, STATS, dbg, dbb, D1H, BATCHC, NOUTC);
  k_gemm_f16<64,0,false,true,false><<<BATCHC/16, 128, 0, stream>>>(D1H, DW2p, db2, nullptr, (float*)d_out, nullptr, NOUTC);
}
